// AttentionBlock_52003464020376
// MI455X (gfx1250) — matrix operand order repaired and checked by reading, not compiled, not run on hardware
//
#include <hip/hip_runtime.h>
#include <cstdint>

// ---------------------------------------------------------------------------
// AttentionBlock on MI455X (gfx1250, wave32, WMMA bf16 16x16x32, f32 accum)
//
// Shapes: x (8,256,32,32) f32; heads=8, hd=32, seq=1024, groups=32.
// Everything runs out of L2 (~35MB working set << 192MB).
//
// Workspace layout (dwords):
//   stats   512            (b,g) mean/rstd
//   wqkvA   98304          w_qkv  A-frags  (48 mt x 8 kt x 256dw)
//   wprojA  32768          w_proj A-frags  (16 mt x 8 kt)
//   xnB     1048576        xn     B-frags  (8b x 8kt x 64nt)
//   qkv     3145728        qkv bf16 row-major (8b x 768 x 1024)
//   Qa      1048576        Q^T A-frags per (b,h): 64 mt x 1 kt  (pre-scaled)
//   Kb      1048576        K   B-frags per (b,h): 1 kt x 64 nt
//   Vb      1048576        V^T B-frags per (b,h): 32 kt x 2 nt
//   attnB   1048576        attn-out B-frags (8b x 8kt x 64nt)
// total = 8,520,192 dw = ~32.5 MB
// ---------------------------------------------------------------------------

typedef __attribute__((ext_vector_type(16))) __bf16 v16bf;
typedef __attribute__((ext_vector_type(8)))  float  v8f;

#define WMMA_BF16(a, b, c) \
  __builtin_amdgcn_wmma_f32_16x16x32_bf16(false, (a), false, (b), (short)0, (c), false, false)

static __device__ __forceinline__ unsigned short f2bf(float f) {
  unsigned u = __float_as_uint(f);
  u += 0x7FFFu + ((u >> 16) & 1u);          // round-to-nearest-even
  return (unsigned short)(u >> 16);
}
static __device__ __forceinline__ float bf2f(unsigned short u) {
  return __uint_as_float(((unsigned)u) << 16);
}
static __device__ __forceinline__ uint32_t pack2bf(float lo, float hi) {
  return (uint32_t)f2bf(lo) | ((uint32_t)f2bf(hi) << 16);
}
// 16-bit A 16x32 — lane l holds M=l&15; vgpr v holds K pair 16*hi+2v (contiguous k order used here):
static __device__ __forceinline__ int afrag_k(int v, int hi) {
  return 16 * hi + 2 * v;
}
// 16-bit B 32x16 — lane l holds N=l&15; lanes 0-15: K=2v, lanes 16-31: K=16+2v
static __device__ __forceinline__ int bfrag_k(int v, int hi) {
  return 16 * hi + 2 * v;
}
// CDNA5 async global->LDS copy (ASYNCcnt); lds_off = addrspace(3) byte offset.
static __device__ __forceinline__ void async_ld_b128(uint32_t lds_off, const void* g) {
  asm volatile("global_load_async_to_lds_b128 %0, %1, off" :: "v"(lds_off), "v"(g) : "memory");
}
static __device__ __forceinline__ void wait_async0() {
  asm volatile("s_wait_asynccnt 0" ::: "memory");
}

// ---------------- 1) GroupNorm stats: mean, rstd per (b, g) ----------------
__global__ void gn_stats(const float* __restrict__ x, float* __restrict__ stats) {
  int b = blockIdx.x, g = blockIdx.y;
  const float* p = x + ((size_t)b * 256 + g * 8) * 1024;
  float s = 0.f, ss = 0.f;
  for (int i = threadIdx.x; i < 8192; i += 256) { float v = p[i]; s += v; ss += v * v; }
  __shared__ float sh[512];
  sh[threadIdx.x] = s; sh[256 + threadIdx.x] = ss;
  __syncthreads();
  for (int st = 128; st > 0; st >>= 1) {
    if (threadIdx.x < st) {
      sh[threadIdx.x] += sh[threadIdx.x + st];
      sh[256 + threadIdx.x] += sh[256 + threadIdx.x + st];
    }
    __syncthreads();
  }
  if (threadIdx.x == 0) {
    float mean = sh[0] * (1.f / 8192.f);
    float var  = sh[256] * (1.f / 8192.f) - mean * mean;
    stats[(b * 32 + g) * 2 + 0] = mean;
    stats[(b * 32 + g) * 2 + 1] = rsqrtf(var + 1e-5f);
  }
}

// ---------------- 2) Pack f32 weight matrix (MxK) into bf16 A-frags --------
__global__ void pack_afrag(const float* __restrict__ W, uint32_t* __restrict__ out,
                           int Mt, int Kt, int K) {
  int idx = blockIdx.x * 256 + threadIdx.x;
  int total = Mt * Kt * 256;
  if (idx >= total) return;
  int tile = idx >> 8, r = idx & 255, lane = r >> 3, v = r & 7;
  int mt = tile / Kt, kt = tile % Kt;
  int m = mt * 16 + (lane & 15), hi = lane >> 4;
  int k = kt * 32 + afrag_k(v, hi);
  const float* row = W + (size_t)m * K + k;
  out[idx] = pack2bf(row[0], row[1]);
}

// ---------------- 3) GroupNorm-on-the-fly, xn -> bf16 B-frags --------------
__global__ void pack_xnb(const float* __restrict__ x, const float* __restrict__ gamma,
                         const float* __restrict__ beta, const float* __restrict__ stats,
                         uint32_t* __restrict__ out) {
  int idx = blockIdx.x * 256 + threadIdx.x;      // 1,048,576 total
  int b = idx >> 17;
  int rem = idx & 131071;
  int tile = rem >> 8, r = rem & 255, lane = r >> 3, v = r & 7;
  int kt = tile >> 6, nt = tile & 63;
  int n = nt * 16 + (lane & 15), hi = lane >> 4;
  int c = kt * 32 + bfrag_k(v, hi);
  int c1 = c + 1;
  float m0 = stats[(b * 32 + (c  >> 3)) * 2], r0 = stats[(b * 32 + (c  >> 3)) * 2 + 1];
  float m1 = stats[(b * 32 + (c1 >> 3)) * 2], r1 = stats[(b * 32 + (c1 >> 3)) * 2 + 1];
  float x0 = x[((size_t)b * 256 + c ) * 1024 + n];
  float x1 = x[((size_t)b * 256 + c1) * 1024 + n];
  float v0 = (x0 - m0) * r0 * gamma[c ] + beta[c ];
  float v1 = (x1 - m1) * r1 * gamma[c1] + beta[c1];
  out[idx] = pack2bf(v0, v1);
}

// ---------------- 4) QKV GEMM: (768x256)x(256x1024) per batch --------------
// Double-buffered k-loop: next A/B frags load while current 4 WMMAs issue.
__global__ void __launch_bounds__(256) qkv_gemm(const uint32_t* __restrict__ wA,
                                                const uint32_t* __restrict__ xnB,
                                                const float* __restrict__ bias,
                                                unsigned short* __restrict__ qkv) {
  int mt = blockIdx.x;           // 0..47
  int nch = blockIdx.y;          // 0..1
  int b = blockIdx.z;
  int w = threadIdx.x >> 5, lane = threadIdx.x & 31;
  int nt0 = nch * 32 + w * 4;    // 4 n-tiles per wave
  const uint32_t* ap = wA + (size_t)mt * 8 * 256 + lane * 8;
  const uint32_t* bp = xnB + (size_t)b * 131072 + (size_t)nt0 * 256 + lane * 8;
  v8f zero = {};
  v8f acc[4] = {zero, zero, zero, zero};
  v16bf a_cur = *(const v16bf*)(ap);
  v16bf b_cur[4];
#pragma unroll
  for (int j = 0; j < 4; ++j) b_cur[j] = *(const v16bf*)(bp + j * 256);
#pragma unroll
  for (int kt = 0; kt < 8; ++kt) {
    v16bf a_nxt = a_cur;
    v16bf b_nxt[4] = {b_cur[0], b_cur[1], b_cur[2], b_cur[3]};
    if (kt < 7) {
      a_nxt = *(const v16bf*)(ap + (size_t)(kt + 1) * 256);
      const uint32_t* bn = bp + (size_t)(kt + 1) * 64 * 256;
#pragma unroll
      for (int j = 0; j < 4; ++j) b_nxt[j] = *(const v16bf*)(bn + j * 256);
    }
#pragma unroll
    for (int j = 0; j < 4; ++j) acc[j] = WMMA_BF16(a_cur, b_cur[j], acc[j]);
    a_cur = a_nxt;
#pragma unroll
    for (int j = 0; j < 4; ++j) b_cur[j] = b_nxt[j];
  }
  int hi = lane >> 4, lo = lane & 15;
#pragma unroll
  for (int j = 0; j < 4; ++j) {
    int n = (nt0 + j) * 16 + lo;
#pragma unroll
    for (int r = 0; r < 8; ++r) {
      int o = mt * 16 + r + 8 * hi;
      qkv[((size_t)b * 768 + o) * 1024 + n] = f2bf(acc[j][r] + bias[o]);
    }
  }
}

// ---------------- 5) Rearrange qkv into attention fragment buffers ---------
__global__ void pack_qkv(const unsigned short* __restrict__ qkv,
                         uint32_t* __restrict__ out, int mode) {
  int idx = blockIdx.x * 256 + threadIdx.x;      // 1,048,576 per mode
  int bh = idx >> 14;                            // 16384 dw per (b,h)
  int b = bh >> 3, h = bh & 7;
  int rem = idx & 16383;
  int tile = rem >> 8, r = rem & 255, lane = r >> 3, v = r & 7;
  int hi = lane >> 4, lo = lane & 15;
  const unsigned short* base = qkv + (size_t)b * 768 * 1024;
  if (mode == 0) {               // Q^T A-frags (64 mt), scale 1/sqrt(hd) folded
    int i = tile * 16 + lo;
    int d = afrag_k(v, hi);
    float f0 = bf2f(base[(size_t)(h * 32 + d    ) * 1024 + i]) * 0.17677669529663687f;
    float f1 = bf2f(base[(size_t)(h * 32 + d + 1) * 1024 + i]) * 0.17677669529663687f;
    out[idx] = pack2bf(f0, f1);
  } else if (mode == 1) {        // K B-frags (64 nt)
    int j = tile * 16 + lo;
    int d = bfrag_k(v, hi);
    unsigned short k0 = base[(size_t)(256 + h * 32 + d    ) * 1024 + j];
    unsigned short k1 = base[(size_t)(256 + h * 32 + d + 1) * 1024 + j];
    out[idx] = (uint32_t)k0 | ((uint32_t)k1 << 16);
  } else {                       // V^T B-frags (32 kt x 2 nt)
    int kt = tile >> 1, nt = tile & 1;
    int d = nt * 16 + lo;
    int j = kt * 32 + bfrag_k(v, hi);      // even -> contiguous pair
    const unsigned short* vrow = base + (size_t)(512 + h * 32 + d) * 1024;
    out[idx] = (uint32_t)vrow[j] | ((uint32_t)vrow[j + 1] << 16);
  }
}

// ---------------- 6) Flash attention: one block per (b,h, i-tile of 128) ---
__global__ void __launch_bounds__(256) attn_kernel(const uint32_t* __restrict__ Qa,
                                                   const uint32_t* __restrict__ Kb,
                                                   const uint32_t* __restrict__ Vb,
                                                   uint32_t* __restrict__ attnB) {
  __shared__ uint32_t ldsK[4096];   // 16 K B-frag tiles (j-window 256)
  __shared__ uint32_t ldsV[4096];   // 16 V^T B-frag tiles
  __shared__ uint32_t ldsS[2048];   // per-wave 256dw transpose scratch
  int bh = blockIdx.x;              // b*8+h
  int ic = blockIdx.y;              // i-chunk (128 rows)
  int b = bh >> 3, h = bh & 7;
  int w = threadIdx.x >> 5, lane = threadIdx.x & 31;
  int hi = lane >> 4, lo = lane & 15;
  const uint32_t* qa_base = Qa + (size_t)bh * 16384;
  const uint32_t* kb_base = Kb + (size_t)bh * 16384;
  const uint32_t* vb_base = Vb + (size_t)bh * 16384;

  v16bf qa = *(const v16bf*)(qa_base + (size_t)(ic * 8 + w) * 256 + lane * 8);
  v8f o0 = {}, o1 = {};
  float mrun[8], lrun[8];
#pragma unroll
  for (int r = 0; r < 8; ++r) { mrun[r] = -1e30f; lrun[r] = 0.f; }

  for (int jp = 0; jp < 4; ++jp) {
    __syncthreads();                      // previous panel fully consumed
    { // async global->LDS panel copy (CDNA5 TDM-lite path, ASYNCcnt tracked)
      const uint32_t* srcK = kb_base + jp * 4096;
      const uint32_t* srcV = vb_base + jp * 4096;
      for (int t = threadIdx.x; t < 1024; t += 256) {
        async_ld_b128((uint32_t)(uintptr_t)(ldsK + t * 4), srcK + t * 4);
        async_ld_b128((uint32_t)(uintptr_t)(ldsV + t * 4), srcV + t * 4);
      }
      wait_async0();
    }
    __syncthreads();                      // panel visible to all waves
#pragma unroll 1
    for (int jt = 0; jt < 8; ++jt) {           // 32-wide j chunk
      v16bf kb0 = *(const v16bf*)(ldsK + (jt * 2 + 0) * 256 + lane * 8);
      v16bf kb1 = *(const v16bf*)(ldsK + (jt * 2 + 1) * 256 + lane * 8);
      v8f z = {};
      v8f s0 = WMMA_BF16(qa, kb0, z);
      v8f s1 = WMMA_BF16(qa, kb1, z);
      // online softmax: row max over this chunk (16-lane-half reduction)
      float cmax[8], al[8];
#pragma unroll
      for (int r = 0; r < 8; ++r) cmax[r] = fmaxf(s0[r], s1[r]);
#pragma unroll
      for (int r = 0; r < 8; ++r) {
        float xv = cmax[r];
        xv = fmaxf(xv, __shfl_xor(xv, 1, 32));
        xv = fmaxf(xv, __shfl_xor(xv, 2, 32));
        xv = fmaxf(xv, __shfl_xor(xv, 4, 32));
        xv = fmaxf(xv, __shfl_xor(xv, 8, 32));
        float mnew = fmaxf(mrun[r], xv);
        al[r] = __expf(mrun[r] - mnew);
        mrun[r] = mnew;
      }
      float p0[8], p1[8];
#pragma unroll
      for (int r = 0; r < 8; ++r) {
        p0[r] = __expf(s0[r] - mrun[r]);
        p1[r] = __expf(s1[r] - mrun[r]);
        lrun[r] = lrun[r] * al[r] + p0[r] + p1[r];
        o0[r] *= al[r];
        o1[r] *= al[r];
      }
      // transpose P (C-layout) -> A-frag via per-wave LDS scratch
      {
        volatile unsigned short* sc = (volatile unsigned short*)(ldsS + w * 256);
#pragma unroll
        for (int r = 0; r < 8; ++r) {
          int row = r + 8 * hi;
          sc[row * 32 + lo]      = f2bf(p0[r]);
          sc[row * 32 + 16 + lo] = f2bf(p1[r]);
        }
        asm volatile("s_wait_dscnt 0" ::: "memory");   // wave-lockstep LDS transpose
        union { uint32_t u[8]; v16bf hv; } pa;
        const volatile uint32_t* scd = (const volatile uint32_t*)(ldsS + w * 256);
#pragma unroll
        for (int vv = 0; vv < 8; ++vv) {
          int k = afrag_k(vv, hi);                     // even -> dword aligned
          pa.u[vv] = scd[lo * 16 + (k >> 1)];
        }
        v16bf vb0 = *(const v16bf*)(ldsV + (jt * 2 + 0) * 256 + lane * 8);
        v16bf vb1 = *(const v16bf*)(ldsV + (jt * 2 + 1) * 256 + lane * 8);
        o0 = WMMA_BF16(pa.hv, vb0, o0);
        o1 = WMMA_BF16(pa.hv, vb1, o1);
      }
    }
  }
  // finalize: row sums and 1/l
#pragma unroll
  for (int r = 0; r < 8; ++r) {
    float xv = lrun[r];
    xv += __shfl_xor(xv, 1, 32);
    xv += __shfl_xor(xv, 2, 32);
    xv += __shfl_xor(xv, 4, 32);
    xv += __shfl_xor(xv, 8, 32);
    float li = 1.0f / xv;
    o0[r] *= li;
    o1[r] *= li;
  }
  // write O as B-frag for the projection GEMM (k-tile = head h)
  {
    volatile unsigned short* sc = (volatile unsigned short*)(ldsS + w * 256);
#pragma unroll
    for (int r = 0; r < 8; ++r) {
      int row = r + 8 * hi;                 // i_local
      sc[row * 32 + lo]      = f2bf(o0[r]); // d = lo
      sc[row * 32 + 16 + lo] = f2bf(o1[r]); // d = 16+lo
    }
    asm volatile("s_wait_dscnt 0" ::: "memory");
    const volatile uint32_t* scd = (const volatile uint32_t*)(ldsS + w * 256);
    uint32_t* outb = attnB + (size_t)b * 131072;
    int nt = ic * 8 + w;
    size_t tbase = ((size_t)h * 64 + nt) * 256;
#pragma unroll
    for (int vv = 0; vv < 8; ++vv) {
      int k = bfrag_k(vv, hi);
      outb[tbase + lane * 8 + vv] = scd[lo * 16 + (k >> 1)];
    }
  }
}

// ---------------- 7) Projection GEMM + bias + residual ---------------------
__global__ void __launch_bounds__(256) proj_gemm(const uint32_t* __restrict__ wA,
                                                 const uint32_t* __restrict__ attnB,
                                                 const float* __restrict__ bias,
                                                 const float* __restrict__ x,
                                                 float* __restrict__ out) {
  int mt = blockIdx.x;           // 0..15
  int nch = blockIdx.y;          // 0..1
  int b = blockIdx.z;
  int w = threadIdx.x >> 5, lane = threadIdx.x & 31;
  int nt0 = nch * 32 + w * 4;
  const uint32_t* ap = wA + (size_t)mt * 8 * 256 + lane * 8;
  const uint32_t* bp = attnB + (size_t)b * 131072 + (size_t)nt0 * 256 + lane * 8;
  v8f zero = {};
  v8f acc[4] = {zero, zero, zero, zero};
  v16bf a_cur = *(const v16bf*)(ap);
  v16bf b_cur[4];
#pragma unroll
  for (int j = 0; j < 4; ++j) b_cur[j] = *(const v16bf*)(bp + j * 256);
#pragma unroll
  for (int kt = 0; kt < 8; ++kt) {
    v16bf a_nxt = a_cur;
    v16bf b_nxt[4] = {b_cur[0], b_cur[1], b_cur[2], b_cur[3]};
    if (kt < 7) {
      a_nxt = *(const v16bf*)(ap + (size_t)(kt + 1) * 256);
      const uint32_t* bn = bp + (size_t)(kt + 1) * 64 * 256;
#pragma unroll
      for (int j = 0; j < 4; ++j) b_nxt[j] = *(const v16bf*)(bn + j * 256);
    }
#pragma unroll
    for (int j = 0; j < 4; ++j) acc[j] = WMMA_BF16(a_cur, b_cur[j], acc[j]);
    a_cur = a_nxt;
#pragma unroll
    for (int j = 0; j < 4; ++j) b_cur[j] = b_nxt[j];
  }
  int hi = lane >> 4, lo = lane & 15;
#pragma unroll
  for (int j = 0; j < 4; ++j) {
    int n = (nt0 + j) * 16 + lo;
#pragma unroll
    for (int r = 0; r < 8; ++r) {
      int o = mt * 16 + r + 8 * hi;
      size_t idx = ((size_t)b * 256 + o) * 1024 + n;
      out[idx] = acc[j][r] + bias[o] + x[idx];
    }
  }
}

// ---------------------------------------------------------------------------
extern "C" void kernel_launch(void* const* d_in, const int* in_sizes, int n_in,
                              void* d_out, int out_size, void* d_ws, size_t ws_size,
                              hipStream_t stream) {
  const float* x      = (const float*)d_in[0];
  const float* gamma  = (const float*)d_in[1];
  const float* beta   = (const float*)d_in[2];
  const float* w_qkv  = (const float*)d_in[3];
  const float* b_qkv  = (const float*)d_in[4];
  const float* w_proj = (const float*)d_in[5];
  const float* b_proj = (const float*)d_in[6];
  float* out = (float*)d_out;

  uint32_t* p = (uint32_t*)d_ws;
  float*    stats  = (float*)p;             p += 512;
  uint32_t* wqkvA  = p;                     p += 98304;
  uint32_t* wprojA = p;                     p += 32768;
  uint32_t* xnB    = p;                     p += 1048576;
  unsigned short* qkv = (unsigned short*)p; p += 3145728;
  uint32_t* Qa     = p;                     p += 1048576;
  uint32_t* Kb     = p;                     p += 1048576;
  uint32_t* Vb     = p;                     p += 1048576;
  uint32_t* attnB  = p;                     p += 1048576;

  gn_stats   <<<dim3(8, 32),    256, 0, stream>>>(x, stats);
  pack_afrag <<<98304 / 256,    256, 0, stream>>>(w_qkv,  wqkvA,  48, 8, 256);
  pack_afrag <<<32768 / 256,    256, 0, stream>>>(w_proj, wprojA, 16, 8, 256);
  pack_xnb   <<<4096,           256, 0, stream>>>(x, gamma, beta, stats, xnB);
  qkv_gemm   <<<dim3(48, 2, 8), 256, 0, stream>>>(wqkvA, xnB, b_qkv, qkv);
  pack_qkv   <<<4096,           256, 0, stream>>>(qkv, Qa, 0);
  pack_qkv   <<<4096,           256, 0, stream>>>(qkv, Kb, 1);
  pack_qkv   <<<4096,           256, 0, stream>>>(qkv, Vb, 2);
  attn_kernel<<<dim3(64, 8),    256, 0, stream>>>(Qa, Kb, Vb, attnB);
  proj_gemm  <<<dim3(16, 2, 8), 256, 0, stream>>>(wprojA, attnB, b_proj, x, out);
}
